// LSTMDecoder_78082505441883
// MI455X (gfx1250) — matrix operand order repaired and checked by reading, not compiled, not run on hardware
//
#include <hip/hip_runtime.h>
#include <math.h>

#define HDIM 1024
#define EDIM 128
#define CDIM 30
#define BDIM 64
#define SDIM 1024
#define TSTEPS 1023            // S-1
#define ROWS (TSTEPS*BDIM)     // 65472
#define G4H 4096
#define IN0 (HDIM+EDIM)        // 1152

typedef __bf16 bf16;
typedef __attribute__((ext_vector_type(16))) __bf16 v16bf;
typedef __attribute__((ext_vector_type(8)))  __bf16 bf16x8;
typedef __attribute__((ext_vector_type(8)))  float  v8f;

// ---------------------------------------------------------------------------
// WMMA fragment helpers (wave32; contiguous k order, shared by A and B)
// A 16x32 bf16: lane holds row m=lane%16; elem j -> K=k0+16*half+j.
//               Two 16B contiguous loads.
// ---------------------------------------------------------------------------
static __device__ __forceinline__ v16bf load_a_frag(const bf16* rowp, int half) {
  bf16x8 lo = *(const bf16x8*)(rowp + 16 * half);
  bf16x8 hi = *(const bf16x8*)(rowp + 16 * half + 8);
  v16bf a;
#pragma unroll
  for (int j = 0; j < 8; ++j) { a[j] = lo[j]; a[j + 8] = hi[j]; }
  return a;
}

// B 32x16 bf16: lane holds col n=lane%16; elem j -> K=k0+16*half+j.
// colp = &W[n*ldk + k0 + 16*half]; 16 contiguous bf16 (32B).
static __device__ __forceinline__ v16bf load_b_frag(const bf16* colp) {
  bf16x8 lo = *(const bf16x8*)(colp);
  bf16x8 hi = *(const bf16x8*)(colp + 8);
  v16bf b;
#pragma unroll
  for (int j = 0; j < 8; ++j) { b[j] = lo[j]; b[j + 8] = hi[j]; }
  return b;
}

static __device__ __forceinline__ v8f wmma_bf16(v16bf a, v16bf b, v8f c) {
  return __builtin_amdgcn_wmma_f32_16x16x32_bf16(false, a, false, b, (short)0, c,
                                                 false, false);
}

static __device__ __forceinline__ float sigf(float x) {
  return 1.0f / (1.0f + __expf(-x));
}

// ---------------------------------------------------------------------------
// Kernel 0: convert weights/activations to bf16, fold biases, zero state.
// ---------------------------------------------------------------------------
__global__ void k_prep(const float* Wih0, const float* Whh0, const float* Wih1,
                       const float* Whh1, const float* bih0, const float* bhh0,
                       const float* bih1, const float* bhh1, const float* cond,
                       const float* emb, bf16* wih0b, bf16* whh0b, bf16* wih1b,
                       bf16* whh1b, float* bias0, float* bias1, bf16* condb,
                       bf16* embb, bf16* h0b, bf16* h1z, float* c0, float* c1,
                       unsigned* barrier) {
  size_t tid = (size_t)blockIdx.x * blockDim.x + threadIdx.x;
  size_t stride = (size_t)gridDim.x * blockDim.x;
  for (size_t i = tid; i < (size_t)G4H * IN0; i += stride) wih0b[i] = (bf16)Wih0[i];
  for (size_t i = tid; i < (size_t)G4H * HDIM; i += stride) {
    whh0b[i] = (bf16)Whh0[i];
    wih1b[i] = (bf16)Wih1[i];
    whh1b[i] = (bf16)Whh1[i];
  }
  for (size_t i = tid; i < G4H; i += stride) {
    bias0[i] = bih0[i] + bhh0[i];
    bias1[i] = bih1[i] + bhh1[i];
  }
  for (size_t i = tid; i < (size_t)BDIM * HDIM; i += stride) {
    condb[i] = (bf16)cond[i];
    c0[i] = 0.0f;
    c1[i] = 0.0f;
    h1z[i] = (bf16)0.0f;
    h0b[i] = (bf16)0.0f;
    h0b[i + (size_t)BDIM * HDIM] = (bf16)0.0f;
  }
  for (size_t i = tid; i < (size_t)CDIM * EDIM; i += stride) embb[i] = (bf16)emb[i];
  if (tid < 2) barrier[tid] = 0u;
}

// ---------------------------------------------------------------------------
// Kernel 1: parallel input GEMM for layer 0.
// G0X[t*64+b, :] = [cond(b) | emb(ids[b,t])] @ Wih0^T + bias0   (bf16 out)
// One wave per 16(M) x 128(N) tile: 1 A frag reused over 8 B frags/WMMAs
// per k-chunk -> 9 fragment loads per 8 WMMAs.
// ---------------------------------------------------------------------------
__global__ __launch_bounds__(256) void k_ingemm(const long long* ids,
                                                const bf16* condb,
                                                const bf16* embb,
                                                const bf16* wih0b,
                                                const float* bias0, bf16* g0x) {
  int wave = blockIdx.x * 8 + (threadIdx.x >> 5);
  const int ntilesN = G4H / 128;           // 32
  int mtile = wave / ntilesN;
  int ntile = wave % ntilesN;
  if (mtile >= ROWS / 16) return;
  int t = mtile >> 2;
  int b0 = (mtile & 3) * 16;
  int n0 = ntile * 128;
  int lane = threadIdx.x & 31;
  int half = lane >> 4;
  int mA = lane & 15;                      // A-fragment row (= batch offset)
  int bA = b0 + mA;
  int id = (int)ids[(size_t)bA * SDIM + t];
  const bf16* crow = condb + (size_t)bA * HDIM;
  const bf16* erow = embb + (size_t)id * EDIM;

  v8f acc[8];
#pragma unroll
  for (int q = 0; q < 8; ++q) acc[q] = (v8f)(0.0f);

  for (int k0 = 0; k0 < IN0; k0 += 32) {
    const bf16* ap = (k0 < HDIM) ? (crow + k0) : (erow + (k0 - HDIM));
    v16bf a = load_a_frag(ap, half);
    const bf16* wb = wih0b + (size_t)(n0 + (lane & 15)) * IN0 + k0 + 16 * half;
#pragma unroll
    for (int q = 0; q < 8; ++q) {
      v16bf bfr = load_b_frag(wb + (size_t)q * 16 * IN0);
      acc[q] = wmma_bf16(a, bfr, acc[q]);
    }
  }
  int row_base = t * 64 + b0;
#pragma unroll
  for (int q = 0; q < 8; ++q) {
    int n = n0 + q * 16 + (lane & 15);
    float bi = bias0[n];
#pragma unroll
    for (int r = 0; r < 8; ++r) {
      int m = r + 8 * half;
      g0x[(size_t)(row_base + m) * G4H + n] = (bf16)(acc[q][r] + bi);
    }
  }
}

// ---------------------------------------------------------------------------
// Kernel 2: persistent recurrent kernel with grid-wide barrier.
// 32 blocks x 8 waves = 256 waves; wave owns a 16(batch) x 16(unit) tile and
// computes all 4 gate tiles -> LSTM cell applied in registers.
// G0X streams from HBM (536 MB > L2), so next step's tile is prefetched
// (global_prefetch_b8, per-lane addresses) while the K-loop runs.
// ---------------------------------------------------------------------------
static __device__ __forceinline__ void grid_sync(unsigned* cnt,
                                                 volatile unsigned* gen,
                                                 unsigned nb) {
  __threadfence();
  __syncthreads();
  if (threadIdx.x == 0) {
    unsigned g = *gen;
    if (atomicAdd(cnt, 1u) == nb - 1u) {
      *cnt = 0u;
      __threadfence();
      atomicAdd((unsigned*)gen, 1u);
    } else {
      while (*gen == g) __builtin_amdgcn_s_sleep(1);
    }
  }
  __syncthreads();
}

static __device__ __forceinline__ void lstm_cell_store(v8f* acc, float* c,
                                                       bf16* hout, int b0,
                                                       int u0, int half,
                                                       int nD) {
#pragma unroll
  for (int r = 0; r < 8; ++r) {
    int b = b0 + r + 8 * half;
    size_t idx = (size_t)b * HDIM + (u0 + nD);
    float ig = sigf(acc[0][r]);
    float fg = sigf(acc[1][r]);
    float gg = tanhf(acc[2][r]);
    float og = sigf(acc[3][r]);
    float cn = fg * c[idx] + ig * gg;
    c[idx] = cn;
    hout[idx] = (bf16)(og * tanhf(cn));
  }
}

__global__ __launch_bounds__(256, 1) void k_recurrent(
    const bf16* g0x, const bf16* whh0b, const bf16* wih1b, const bf16* whh1b,
    const float* bias1, bf16* h0b, const bf16* h1z, bf16* ys, float* c0,
    float* c1, unsigned* barrier) {
  const unsigned NB = gridDim.x;
  unsigned* cnt = barrier;
  volatile unsigned* gen = barrier + 1;
  int wave = blockIdx.x * 8 + (threadIdx.x >> 5);  // 0..255
  int lane = threadIdx.x & 31;
  int half = lane >> 4;
  int mA = lane & 15;
  int nD = lane & 15;
  int b0 = (wave >> 6) * 16;                       // 4 batch tiles
  int u0 = (wave & 63) * 16;                       // 64 unit tiles
  const size_t HB = (size_t)BDIM * HDIM;

  for (int t = 0; t < TSTEPS; ++t) {
    // ---------------- layer 0: gates = G0X[t] + h0_prev @ Whh0^T -----------
    {
      const bf16* hp = h0b + (size_t)(t & 1) * HB + (size_t)(b0 + mA) * HDIM;
      bf16* hw = h0b + (size_t)((t & 1) ^ 1) * HB;
      const bf16* gx = g0x + (size_t)(t * 64 + b0) * G4H;
      v8f acc[4];
#pragma unroll
      for (int g = 0; g < 4; ++g)
#pragma unroll
        for (int r = 0; r < 8; ++r)
          acc[g][r] = (float)gx[(size_t)(r + 8 * half) * G4H + g * HDIM + u0 + nD];
      // Prefetch next timestep's G0X tile (16 rows x 4 gate segments);
      // lane L covers row (L&15), gate (L>>4) then gate (L>>4)+2.
      if (t + 1 < TSTEPS) {
        const bf16* gxn =
            g0x + (size_t)((t + 1) * 64 + b0 + (lane & 15)) * G4H + u0;
        __builtin_prefetch(gxn + (size_t)(lane >> 4) * HDIM, 0, 3);
        __builtin_prefetch(gxn + (size_t)((lane >> 4) + 2) * HDIM, 0, 3);
      }
      for (int k0 = 0; k0 < HDIM; k0 += 32) {
        v16bf a = load_a_frag(hp + k0, half);
#pragma unroll
        for (int g = 0; g < 4; ++g) {
          const bf16* wp =
              whh0b + (size_t)(g * HDIM + u0 + nD) * HDIM + k0 + 16 * half;
          acc[g] = wmma_bf16(a, load_b_frag(wp), acc[g]);
        }
      }
      lstm_cell_store(acc, c0, hw, b0, u0, half, nD);
    }
    grid_sync(cnt, gen, NB);
    // ------- layer 1: gates = h0_t @ Wih1^T + h1_prev @ Whh1^T + bias1 -----
    {
      const bf16* x0 = h0b + (size_t)((t & 1) ^ 1) * HB + (size_t)(b0 + mA) * HDIM;
      const bf16* h1p = (t == 0)
                            ? (h1z + (size_t)(b0 + mA) * HDIM)
                            : (ys + ((size_t)(t - 1) * BDIM + b0 + mA) * HDIM);
      v8f acc[4];
#pragma unroll
      for (int g = 0; g < 4; ++g) {
        float bi = bias1[g * HDIM + u0 + nD];
#pragma unroll
        for (int r = 0; r < 8; ++r) acc[g][r] = bi;
      }
      for (int k0 = 0; k0 < HDIM; k0 += 32) {
        v16bf a0 = load_a_frag(x0 + k0, half);
        v16bf a1 = load_a_frag(h1p + k0, half);
#pragma unroll
        for (int g = 0; g < 4; ++g) {
          size_t wo = (size_t)(g * HDIM + u0 + nD) * HDIM + k0 + 16 * half;
          acc[g] = wmma_bf16(a0, load_b_frag(wih1b + wo), acc[g]);
          acc[g] = wmma_bf16(a1, load_b_frag(whh1b + wo), acc[g]);
        }
      }
      lstm_cell_store(acc, c1, ys + (size_t)t * HB, b0, u0, half, nD);
    }
    grid_sync(cnt, gen, NB);
  }
}

// ---------------------------------------------------------------------------
// Kernel 3: logits over C=30, log-softmax, NLL; deterministic per-block sums.
// ---------------------------------------------------------------------------
__global__ __launch_bounds__(256) void k_loss(const bf16* ys, const float* fcw,
                                              const float* fcb,
                                              const long long* ids,
                                              const float* attn,
                                              float* partials) {
  int wv = blockIdx.x * 8 + (threadIdx.x >> 5);
  int lane = threadIdx.x & 31;
  int nwaves = gridDim.x * 8;
  float s_nll = 0.0f, s_mask = 0.0f;
  for (int row = wv; row < ROWS; row += nwaves) {
    int t = row >> 6, b = row & 63;
    const bf16* h = ys + (size_t)row * HDIM;
    float la[CDIM];
#pragma unroll
    for (int c = 0; c < CDIM; ++c) la[c] = 0.0f;
    for (int k = lane; k < HDIM; k += 32) {
      float hv = (float)h[k];
#pragma unroll
      for (int c = 0; c < CDIM; ++c) la[c] += hv * fcw[c * HDIM + k];
    }
#pragma unroll
    for (int c = 0; c < CDIM; ++c) {
      float v = la[c];
      for (int off = 16; off > 0; off >>= 1) v += __shfl_xor(v, off, 32);
      la[c] = v + fcb[c];
    }
    int tgt = (int)ids[(size_t)b * SDIM + (t + 1)];
    float mx = la[0], lt = la[0];
#pragma unroll
    for (int c = 1; c < CDIM; ++c) {
      mx = fmaxf(mx, la[c]);
      if (c == tgt) lt = la[c];
    }
    if (tgt == 0) lt = la[0];
    float se = 0.0f;
#pragma unroll
    for (int c = 0; c < CDIM; ++c) se += __expf(la[c] - mx);
    float nll = (mx + __logf(se)) - lt;
    s_nll += nll;
    s_mask += attn[(size_t)b * SDIM + (t + 1)];
  }
  __shared__ float red[8][2];
  if (lane == 0) {
    red[threadIdx.x >> 5][0] = s_nll;
    red[threadIdx.x >> 5][1] = s_mask;
  }
  __syncthreads();
  if (threadIdx.x == 0) {
    float a = 0.0f, m = 0.0f;
    for (int w = 0; w < 8; ++w) { a += red[w][0]; m += red[w][1]; }
    partials[blockIdx.x * 2 + 0] = a;
    partials[blockIdx.x * 2 + 1] = m;
  }
}

__global__ void k_final(const float* partials, int nparts, float* out) {
  if (threadIdx.x == 0 && blockIdx.x == 0) {
    float sn = 0.0f, sm = 0.0f;
    for (int i = 0; i < nparts; ++i) {
      sn += partials[2 * i + 0];
      sm += partials[2 * i + 1];
    }
    float ce = sn / (float)ROWS;
    float masked = (sm != 0.0f) ? (ce * sm) / sm : 0.0f;
    out[0] = ce + masked;
  }
}

// ---------------------------------------------------------------------------
// Host launcher
// ---------------------------------------------------------------------------
static inline size_t align256(size_t x) { return (x + 255) & ~(size_t)255; }

extern "C" void kernel_launch(void* const* d_in, const int* in_sizes, int n_in,
                              void* d_out, int out_size, void* d_ws,
                              size_t ws_size, hipStream_t stream) {
  const long long* ids = (const long long*)d_in[0];
  const float* attn = (const float*)d_in[1];
  const float* cond = (const float*)d_in[2];
  const float* emb = (const float*)d_in[3];
  const float* Wih0 = (const float*)d_in[4];
  const float* Whh0 = (const float*)d_in[5];
  const float* bih0 = (const float*)d_in[6];
  const float* bhh0 = (const float*)d_in[7];
  const float* Wih1 = (const float*)d_in[8];
  const float* Whh1 = (const float*)d_in[9];
  const float* bih1 = (const float*)d_in[10];
  const float* bhh1 = (const float*)d_in[11];
  const float* fcw = (const float*)d_in[12];
  const float* fcb = (const float*)d_in[13];
  float* out = (float*)d_out;

  char* ws = (char*)d_ws;
  size_t off = 0;
  auto alloc = [&](size_t bytes) -> char* {
    char* p = ws + off;
    off = align256(off + bytes);
    return p;
  };
  unsigned* barrier = (unsigned*)alloc(2 * sizeof(unsigned));
  float* partials = (float*)alloc(256 * 2 * sizeof(float));
  float* bias0 = (float*)alloc((size_t)G4H * 4);
  float* bias1 = (float*)alloc((size_t)G4H * 4);
  bf16* condb = (bf16*)alloc((size_t)BDIM * HDIM * 2);
  bf16* embb = (bf16*)alloc((size_t)CDIM * EDIM * 2);
  bf16* wih0b = (bf16*)alloc((size_t)G4H * IN0 * 2);
  bf16* whh0b = (bf16*)alloc((size_t)G4H * HDIM * 2);
  bf16* wih1b = (bf16*)alloc((size_t)G4H * HDIM * 2);
  bf16* whh1b = (bf16*)alloc((size_t)G4H * HDIM * 2);
  bf16* h0b = (bf16*)alloc((size_t)2 * BDIM * HDIM * 2);
  bf16* h1z = (bf16*)alloc((size_t)BDIM * HDIM * 2);
  float* c0 = (float*)alloc((size_t)BDIM * HDIM * 4);
  float* c1 = (float*)alloc((size_t)BDIM * HDIM * 4);
  bf16* g0x = (bf16*)alloc((size_t)ROWS * G4H * 2);
  bf16* ysb = (bf16*)alloc((size_t)ROWS * HDIM * 2);

  k_prep<<<1024, 256, 0, stream>>>(Wih0, Whh0, Wih1, Whh1, bih0, bhh0, bih1,
                                   bhh1, cond, emb, wih0b, whh0b, wih1b, whh1b,
                                   bias0, bias1, condb, embb, h0b, h1z, c0, c1,
                                   barrier);
  // (ROWS/16) * (4096/128) wave-tiles / 8 waves per block = 16368 blocks
  k_ingemm<<<(ROWS / 16) * (G4H / 128) / 8, 256, 0, stream>>>(
      ids, condb, embb, wih0b, bias0, g0x);
  k_recurrent<<<32, 256, 0, stream>>>(g0x, whh0b, wih1b, whh1b, bias1, h0b,
                                      h1z, ysb, c0, c1, barrier);
  k_loss<<<256, 256, 0, stream>>>(ysb, fcw, fcb, ids, attn, partials);
  k_final<<<1, 64, 0, stream>>>(partials, 256, out);
}
